// SparseMoE_70257075028650
// MI455X (gfx1250) — matrix operand order repaired and checked by reading, not compiled, not run on hardware
//
#include <hip/hip_runtime.h>
#include <hip/hip_bf16.h>

typedef __attribute__((ext_vector_type(16))) __bf16 v16bf;
typedef __attribute__((ext_vector_type(8)))  float  v8f;

#define D_      1024
#define H_      4096
#define E_      8
#define NTOK    4096          // B*S
#define TILE_M  32            // tokens per expert tile
#define KS      128           // phase-1 K stage width (4 WMMAs per barrier pair)
#define HC      64            // h-chunk width (phase-1 output / phase-2 K)
#define NC      128           // phase-2 N chunk
#define TILES_MAX (NTOK / TILE_M)

// ---------------------------------------------------------------- zero kernel
__global__ __launch_bounds__(256) void moe_zero(float* __restrict__ out,
                                                int* __restrict__ counts) {
  size_t gid = (size_t)blockIdx.x * blockDim.x + threadIdx.x;
  size_t total = (size_t)NTOK * D_;
  for (size_t i = gid; i < total; i += (size_t)gridDim.x * blockDim.x) out[i] = 0.f;
  if (gid < E_) counts[gid] = 0;
}

// ---------------------------------------------------------------- router
// one block (256 thr) per token: 32 partial lanes per expert, LDS reduce,
// top-2 + softmax over the two selected scores, append to per-expert lists.
__global__ __launch_bounds__(256) void moe_router(
    const float* __restrict__ x, const float* __restrict__ Wr,
    const float* __restrict__ br, int* __restrict__ counts,
    int* __restrict__ tok_list, float* __restrict__ w_list) {
  int token = blockIdx.x;
  int tid   = threadIdx.x;
  int e     = tid & 7;        // expert
  int part  = tid >> 3;       // 0..31
  const float* xrow = x + (size_t)token * D_;
  float partial = 0.f;
  int base = part * (D_ / 32);
  for (int j = 0; j < D_ / 32; ++j) {
    int d = base + j;
    partial += xrow[d] * Wr[(size_t)d * E_ + e];
  }
  __shared__ float red[256];
  red[part * 8 + e] = partial;
  __syncthreads();
  for (int s = 16; s >= 1; s >>= 1) {
    if (part < s) red[part * 8 + e] += red[(part + s) * 8 + e];
    __syncthreads();
  }
  if (tid == 0) {
    float sc[E_];
    for (int i = 0; i < E_; ++i) sc[i] = red[i] + br[i];
    int i1 = 0;
    for (int i = 1; i < E_; ++i) if (sc[i] > sc[i1]) i1 = i;          // argmax (first on tie)
    int i2 = (i1 == 0) ? 1 : 0;
    for (int i = 0; i < E_; ++i) if (i != i1 && sc[i] > sc[i2]) i2 = i;
    float m  = fmaxf(sc[i1], sc[i2]);
    float ea = __expf(sc[i1] - m), eb = __expf(sc[i2] - m);
    float inv = 1.f / (ea + eb);
    int p1 = atomicAdd(&counts[i1], 1);
    tok_list[i1 * NTOK + p1] = token;  w_list[i1 * NTOK + p1] = ea * inv;
    int p2 = atomicAdd(&counts[i2], 1);
    tok_list[i2 * NTOK + p2] = token;  w_list[i2 * NTOK + p2] = eb * inv;
  }
}

// ---------------------------------------------------------------- fused expert FFN
// block = (expert, 32-token tile), 256 threads = 8 waves (wave32).
// phase 1: h[32][64] = relu(x_tile @ W1[:,h0:h0+64] + b1)  (1 WMMA tile / wave,
//          K staged 128 wide -> 4 WMMAs per barrier pair)
// phase 2: y[32][1024] += h @ W2[h0:h0+64,:]  (16 persistent v8f accs / wave)
__global__ __launch_bounds__(256) void moe_expert(
    const float* __restrict__ x,
    const float* __restrict__ W1, const float* __restrict__ b1,
    const float* __restrict__ W2, const float* __restrict__ b2,
    const int* __restrict__ counts, const int* __restrict__ tok_list,
    const float* __restrict__ w_list, float* __restrict__ out) {
  int e    = blockIdx.x / TILES_MAX;
  int tile = blockIdx.x % TILES_MAX;
  int cnt  = counts[e];
  if (tile * TILE_M >= cnt) return;   // block-uniform early exit

  __shared__ __bf16 xa  [TILE_M][KS + 8];   // x stage  [m][k]        8.5 KB
  __shared__ __bf16 w1s [HC]    [KS + 8];   // W1 stage [h][k] (T)   17.0 KB
  __shared__ __bf16 hbuf[TILE_M][HC + 8];   // relu(h)  [m][h]        4.5 KB
  __shared__ __bf16 w2s [NC]    [HC + 8];   // W2 stage [n][h] (T)   18.0 KB
  __shared__ int    toks[TILE_M];
  __shared__ float  wts [TILE_M];

  int tid    = threadIdx.x;
  int waveId = tid >> 5;
  int lane   = tid & 31;
  bool lo    = lane < 16;
  int  l15   = lane & 15;

  if (tid < TILE_M) {
    int idx = tile * TILE_M + tid;
    toks[tid] = (idx < cnt) ? tok_list[e * NTOK + idx] : 0;
    wts [tid] = (idx < cnt) ? w_list [e * NTOK + idx] : 0.f;
  }
  __syncthreads();

  // phase-1 tile ownership: 2 M-tiles x 4 N-tiles = 8 tiles = 8 waves
  int mi = waveId >> 2;     // 0..1
  int ni = waveId & 3;      // 0..3
  // phase-2 tile ownership: 2 M-tiles x 8 N-tiles per n-chunk; wave owns 2
  int mi2    = waveId & 1;  // 0..1
  int njbase = waveId >> 1; // 0..3 -> n-tiles njbase and njbase+4

  v8f acc2[8][2];           // persistent y accumulators: [n-chunk][tile]
  #pragma unroll
  for (int c = 0; c < 8; ++c)
    #pragma unroll
    for (int t = 0; t < 2; ++t)
      #pragma unroll
      for (int r = 0; r < 8; ++r) acc2[c][t][r] = 0.f;

  const size_t w1base = (size_t)e * D_ * H_;
  const size_t w2base = (size_t)e * H_ * D_;

  for (int h0 = 0; h0 < H_; h0 += HC) {
    // ---------------- phase 1: h = relu(x @ W1[:, h0:h0+HC] + b1) ----------
    v8f acc = {0.f, 0.f, 0.f, 0.f, 0.f, 0.f, 0.f, 0.f};
    for (int k0 = 0; k0 < D_; k0 += KS) {
      __syncthreads();  // prior readers of xa/w1s done
      // stage x chunk (gathered rows, fp32 -> bf16), coalesced 128-wide rows
      for (int i = tid; i < TILE_M * KS; i += 256) {
        int r = i >> 7, kk = i & (KS - 1);
        bool rv = (tile * TILE_M + r) < cnt;
        xa[r][kk] = rv ? (__bf16)x[(size_t)toks[r] * D_ + k0 + kk] : (__bf16)0.f;
      }
      // stage W1 chunk transposed: w1s[h][k]; global-coalesced along h (H-major)
      for (int i = tid; i < HC * KS; i += 256) {
        int kk = i >> 6, hh = i & 63;
        w1s[hh][kk] = (__bf16)W1[w1base + (size_t)(k0 + kk) * H_ + h0 + hh];
      }
      if (k0 + KS < D_)  // warm L2 for next chunk
        __builtin_prefetch(&W1[w1base + (size_t)(k0 + KS + (tid >> 6)) * H_ + h0], 0, 1);
      __syncthreads();
      // 4 WMMAs per stage; fragments per documented CDNA5 bf16 layouts
      #pragma unroll
      for (int kq = 0; kq < KS; kq += 32) {
        v16bf a, b;
        int m   = mi * 16 + l15;
        int kA0 = kq + (lo ? 0 : 8), kA1 = kq + (lo ? 16 : 24);
        #pragma unroll
        for (int i = 0; i < 8; ++i) { a[i] = xa[m][kA0 + i]; a[8 + i] = xa[m][kA1 + i]; }
        int n  = ni * 16 + l15;
        int kB0 = kq + (lo ? 0 : 8), kB1 = kq + (lo ? 16 : 24);
        #pragma unroll
        for (int i = 0; i < 8; ++i) { b[i] = w1s[n][kB0 + i]; b[8 + i] = w1s[n][kB1 + i]; }
        acc = __builtin_amdgcn_wmma_f32_16x16x32_bf16(false, a, false, b,
                                                      (short)0, acc, false, false);
      }
    }
    __syncthreads();
    // bias + relu -> hbuf (each wave writes its exclusive 16x16 tile)
    #pragma unroll
    for (int r = 0; r < 8; ++r) {
      int M  = mi * 16 + (lo ? 0 : 8) + r;
      int Nc = ni * 16 + l15;
      float v = acc[r] + b1[e * H_ + h0 + Nc];
      hbuf[M][Nc] = (__bf16)fmaxf(v, 0.f);
    }
    __syncthreads();

    // ---------------- phase 2: y += h @ W2[h0:h0+HC, :] --------------------
    for (int c = 0; c < 8; ++c) {
      int n0 = c * NC;
      // stage W2 chunk transposed: w2s[n][k]; global-coalesced along n (D-major)
      for (int i = tid; i < NC * HC; i += 256) {
        int kk = i >> 7, nn = i & 127;
        w2s[nn][kk] = (__bf16)W2[w2base + (size_t)(h0 + kk) * D_ + n0 + nn];
      }
      __syncthreads();
      #pragma unroll
      for (int t = 0; t < 2; ++t) {
        int nj = njbase + t * 4;
        #pragma unroll
        for (int kk = 0; kk < HC; kk += 32) {
          v16bf a2, bv;
          int m   = mi2 * 16 + l15;
          int kA0 = kk + (lo ? 0 : 8), kA1 = kk + (lo ? 16 : 24);
          #pragma unroll
          for (int i = 0; i < 8; ++i) { a2[i] = hbuf[m][kA0 + i]; a2[8 + i] = hbuf[m][kA1 + i]; }
          int n  = nj * 16 + l15;
          int kB0 = kk + (lo ? 0 : 8), kB1 = kk + (lo ? 16 : 24);
          #pragma unroll
          for (int i = 0; i < 8; ++i) { bv[i] = w2s[n][kB0 + i]; bv[8 + i] = w2s[n][kB1 + i]; }
          acc2[c][t] = __builtin_amdgcn_wmma_f32_16x16x32_bf16(false, a2, false, bv,
                                                               (short)0, acc2[c][t], false, false);
        }
      }
      __syncthreads();  // chunk reads done before restage / next-h0 hbuf write
    }
  }

  // ---------------- epilogue: +b2, * routing weight, scatter-add ----------
  #pragma unroll
  for (int c = 0; c < 8; ++c) {
    #pragma unroll
    for (int t = 0; t < 2; ++t) {
      int nj = njbase + t * 4;
      #pragma unroll
      for (int r = 0; r < 8; ++r) {
        int M  = mi2 * 16 + (lo ? 0 : 8) + r;
        int Nc = c * NC + nj * 16 + l15;
        float v = (acc2[c][t][r] + b2[e * D_ + Nc]) * wts[M];
        atomicAdd(&out[(size_t)toks[M] * D_ + Nc], v);
      }
    }
  }
}

// ---------------------------------------------------------------- launcher
extern "C" void kernel_launch(void* const* d_in, const int* in_sizes, int n_in,
                              void* d_out, int out_size, void* d_ws, size_t ws_size,
                              hipStream_t stream) {
  const float* x  = (const float*)d_in[0];
  const float* Wr = (const float*)d_in[1];
  const float* br = (const float*)d_in[2];
  const float* W1 = (const float*)d_in[3];
  const float* b1 = (const float*)d_in[4];
  const float* W2 = (const float*)d_in[5];
  const float* b2 = (const float*)d_in[6];
  float* out = (float*)d_out;

  char* ws = (char*)d_ws;
  int*   counts   = (int*)ws;                                      // 8 ints
  int*   tok_list = (int*)(ws + 256);                              // E*NTOK ints
  float* w_list   = (float*)(ws + 256 + sizeof(int) * E_ * NTOK);  // E*NTOK floats

  moe_zero  <<<1024, 256, 0, stream>>>(out, counts);
  moe_router<<<NTOK, 256, 0, stream>>>(x, Wr, br, counts, tok_list, w_list);
  moe_expert<<<E_ * TILES_MAX, 256, 0, stream>>>(x, W1, b1, W2, b2,
                                                 counts, tok_list, w_list, out);
}
